// TransformerBlock_40286793236984
// MI455X (gfx1250) — matrix operand order repaired and checked by reading, not compiled, not run on hardware
//
#include <hip/hip_runtime.h>
#include <hip/hip_bf16.h>
#include <math.h>

typedef __bf16 bf16t;
typedef __attribute__((ext_vector_type(16))) __bf16 v16bf;
typedef __attribute__((ext_vector_type(8)))  __bf16 v8bf;
typedef __attribute__((ext_vector_type(8)))  float  v8f;

#define B_      2
#define L_      2048
#define D_      1024
#define H_      16
#define HD_     64
#define W_      128
#define RATIO_  8
#define STRIDE_ 4
#define LC_     511
#define LCP_    512   // padded L for transposed V_c (16B-aligned fragment rows)
#define NEGINF  (-1.0e30f)

union ABFrag { v16bf v; v8bf h[2]; bf16t e[16]; };

static __device__ __forceinline__ v8f wmma_bf16(const ABFrag& a, const ABFrag& b, v8f c) {
  return __builtin_amdgcn_wmma_f32_16x16x32_bf16(false, a.v, false, b.v, (short)0, c, false, false);
}

// round-to-nearest-even f32 -> bf16
static __device__ __forceinline__ bf16t f2b(float f) {
  union { float f; unsigned u; } a; a.f = f;
  unsigned r = a.u + 0x7FFFu + ((a.u >> 16) & 1u);
  union { unsigned short s; bf16t b; } o; o.s = (unsigned short)(r >> 16);
  return o.b;
}

// LDS byte offset of a __shared__ object = low 32 bits of its flat address (ISA 10.2)
static __device__ __forceinline__ unsigned lds_off(const void* p) {
  return (unsigned)(size_t)p;
}
static __device__ __forceinline__ void async_g2l_b128(unsigned l, const bf16t* g) {
  asm volatile("global_load_async_to_lds_b128 %0, %1, off" :: "v"(l), "v"(g) : "memory");
}
static __device__ __forceinline__ void wait_async_le9() {
  asm volatile("s_wait_asynccnt 0x9" ::: "memory");
}
static __device__ __forceinline__ void wait_async_0() {
  asm volatile("s_wait_asynccnt 0x0" ::: "memory");
}

// ---------------- elementwise conversions ----------------

__global__ void cvt_bf16_kernel(const float* __restrict__ in, bf16t* __restrict__ out, int n) {
  int tid = blockIdx.x * blockDim.x + threadIdx.x;
  if (tid < n) out[tid] = f2b(in[tid]);
}

// in: f32 [B, L, H*HD]  ->  out: bf16 [B, H, L, HD] with RoPE applied
__global__ void rope_transpose_kernel(const float* __restrict__ in, bf16t* __restrict__ out) {
  int tid = blockIdx.x * blockDim.x + threadIdx.x;
  if (tid >= B_ * L_ * H_ * (HD_ / 2)) return;
  int j = tid & 31;
  int h = (tid >> 5) & (H_ - 1);
  int l = (tid >> 9) & (L_ - 1);
  int b = tid >> 20;
  float inv = __expf(-(float)j * (9.210340372f / 32.0f)); // 1/10000^(j/32)
  float ang = (float)l * inv;
  float s, c;
  __sincosf(ang, &s, &c);
  size_t ib = ((size_t)(b * L_ + l)) * D_ + (size_t)h * HD_;
  float x1 = in[ib + j];
  float x2 = in[ib + 32 + j];
  size_t ob = (((size_t)(b * H_ + h)) * L_ + l) * HD_;
  out[ob + j]      = f2b(x1 * c - x2 * s);
  out[ob + 32 + j] = f2b(x1 * s + x2 * c);
}

// in: f32 [B, Lx, H*HD] -> out: bf16 [B, H, Lx, HD]   (row-major heads, for K)
__global__ void cvt_transpose_kernel(const float* __restrict__ in, bf16t* __restrict__ out, int Lx) {
  int tid = blockIdx.x * blockDim.x + threadIdx.x;
  if (tid >= B_ * Lx * D_) return;
  int d = tid & 63;
  int h = (tid >> 6) & (H_ - 1);
  int l = (tid >> 10) % Lx;
  int b = tid / (Lx * D_);
  out[(((size_t)(b * H_ + h)) * Lx + l) * HD_ + d] = f2b(in[tid]);
}

// in: f32 [B, Lx, H*HD] -> out: bf16 [B, H, HD, LP] (dim-major V for contiguous PV B-fragments)
// pads l in [Lx, LP) with zeros so fragment loads never touch garbage.
__global__ void cvt_transpose_vT_kernel(const float* __restrict__ in, bf16t* __restrict__ out,
                                        int Lx, int LP) {
  int tid = blockIdx.x * blockDim.x + threadIdx.x;
  if (tid >= B_ * H_ * HD_ * LP) return;
  int l = tid % LP;
  int d = (tid / LP) & 63;
  int h = (tid / (LP * 64)) & (H_ - 1);
  int b = tid / (LP * 64 * H_);
  float v = 0.f;
  if (l < Lx) v = in[(((size_t)b * Lx + l) * D_) + h * 64 + d];
  out[tid] = f2b(v);
}

// gate-softmax-weighted window compression
__global__ void compress_kernel(const float* __restrict__ x, const float* __restrict__ gate,
                                bf16t* __restrict__ xc) {
  int tid = blockIdx.x * blockDim.x + threadIdx.x;
  if (tid >= B_ * LC_ * D_) return;
  int d  = tid & (D_ - 1);
  int wl = (tid >> 10) % LC_;
  int b  = tid / (LC_ * D_);
  float g[RATIO_], mx = NEGINF, ss = 0.f;
#pragma unroll
  for (int r = 0; r < RATIO_; ++r) { g[r] = gate[r]; mx = fmaxf(mx, g[r]); }
#pragma unroll
  for (int r = 0; r < RATIO_; ++r) { g[r] = __expf(g[r] - mx); ss += g[r]; }
  float invs = 1.f / ss, acc = 0.f;
#pragma unroll
  for (int r = 0; r < RATIO_; ++r)
    acc += g[r] * invs * x[((size_t)(b * L_ + wl * STRIDE_ + r)) * D_ + d];
  xc[tid] = f2b(acc);
}

// ---------------- WMMA GEMM: C[M,N] = A[M,K] * B[N,K]^T  (bf16 in, f32 out) ----------------
// 128 threads = 4 waves (2M x 2N); block tile 32x256, wave tile 16x128, K-step 32.
// LDS double-buffered, filled via global_load_async_to_lds (ASYNCcnt path).
__global__ __launch_bounds__(128) void gemm_nt_bf16(
    const bf16t* __restrict__ A, const bf16t* __restrict__ Bw,
    float* __restrict__ C, int M, int N, int K) {
  __shared__ __align__(16) bf16t Asm[2][32][32];    //  4 KB
  __shared__ __align__(16) bf16t Bsm[2][256][32];   // 32 KB
  const int tid  = threadIdx.x;
  const int lane = tid & 31, wave = tid >> 5;
  const int waveM = wave >> 1, waveN = wave & 1;
  const int half = lane >> 4, ncol = lane & 15;
  const int m0 = blockIdx.y * 32;
  const int n0 = blockIdx.x * 256;

  // uniform 9 async ops per thread per stage: 8x b128 (two B rows) + 1x b128 (A piece)
  auto fill = [&](int buf, int k0) {
#pragma unroll
    for (int rr = 0; rr < 2; ++rr) {
      int row = tid + rr * 128;
      const bf16t* gB = Bw + (size_t)(n0 + row) * K + k0;
      unsigned lB = lds_off(&Bsm[buf][row][0]);
      async_g2l_b128(lB,      gB);
      async_g2l_b128(lB + 16, gB + 8);
      async_g2l_b128(lB + 32, gB + 16);
      async_g2l_b128(lB + 48, gB + 24);
    }
    int row = tid >> 2, part = tid & 3;   // 32 rows x 4 pieces of 8 elems
    int arow = m0 + row; if (arow >= M) arow = M - 1;
    async_g2l_b128(lds_off(&Asm[buf][row][part * 8]),
                   A + (size_t)arow * K + k0 + part * 8);
  };

  const v8f vzero = {0.f,0.f,0.f,0.f,0.f,0.f,0.f,0.f};
  v8f acc[8] = {vzero, vzero, vzero, vzero, vzero, vzero, vzero, vzero};

  // register-stage every fragment for a K-step, then 8 back-to-back WMMAs
  auto compute = [&](int buf) {
    ABFrag a0;
    const bf16t* ap = &Asm[buf][waveM * 16 + ncol][0];
    a0.h[0] = *(const v8bf*)(ap + half * 8);
    a0.h[1] = *(const v8bf*)(ap + 16 + half * 8);
    ABFrag bfr[8];
#pragma unroll
    for (int nc = 0; nc < 8; ++nc) {
      const bf16t* bp = &Bsm[buf][waveN * 128 + nc * 16 + ncol][half * 8];
      bfr[nc].h[0] = *(const v8bf*)bp;
      bfr[nc].h[1] = *(const v8bf*)(bp + 16);
    }
#pragma unroll
    for (int nc = 0; nc < 8; ++nc) acc[nc] = wmma_bf16(a0, bfr[nc], acc[nc]);
  };

  const int nk = K >> 5;
  fill(0, 0);
  for (int ks = 0; ks < nk; ks += 2) {
    const bool more1 = (ks + 1 < nk);
    if (more1) { fill(1, (ks + 1) << 5); wait_async_le9(); }
    else       { wait_async_0(); }
    __syncthreads();
    compute(0);
    __syncthreads();
    if (!more1) break;
    const bool more2 = (ks + 2 < nk);
    if (more2) { fill(0, (ks + 2) << 5); wait_async_le9(); }
    else       { wait_async_0(); }
    __syncthreads();
    compute(1);
    __syncthreads();
  }
#pragma unroll
  for (int nc = 0; nc < 8; ++nc)
#pragma unroll
    for (int r = 0; r < 8; ++r) {
      int row = m0 + waveM * 16 + r + half * 8;
      if (row < M) C[(size_t)row * N + (n0 + waveN * 128 + nc * 16 + ncol)] = acc[nc][r];
    }
}

// ---------------- flash attention over [compressed keys | sliding window | sink] ----------------
// K buffers: [B,H,L,HD] row-major. V buffers: [B,H,HD,LP] dim-major (transposed).
__global__ __launch_bounds__(128) void attn_kernel(
    const bf16t* __restrict__ qb, const bf16t* __restrict__ kbuf,
    const bf16t* __restrict__ vbuf, const bf16t* __restrict__ kcb,
    const bf16t* __restrict__ vcb, const float* __restrict__ sinkl,
    bf16t* __restrict__ outb) {
  __shared__ __align__(16) bf16t pshare[4][16][32];
  const int lane = threadIdx.x & 31, wave = threadIdx.x >> 5;
  const int half = lane >> 4, ncol = lane & 15;
  const int tiles_per_bh = L_ / 64;
  const int bh   = blockIdx.x / tiles_per_bh;
  const int qblk = blockIdx.x % tiles_per_bh;
  const int h = bh & (H_ - 1);
  const int qs = qblk * 64 + wave * 16;
  const bf16t* Q   = qb   + (size_t)bh * L_  * HD_;
  const bf16t* Kw  = kbuf + (size_t)bh * L_  * HD_;
  const bf16t* VwT = vbuf + (size_t)bh * HD_ * L_;
  const bf16t* Kc  = kcb  + (size_t)bh * LC_ * HD_;
  const bf16t* VcT = vcb  + (size_t)bh * HD_ * LCP_;

  ABFrag qa0, qa1;
  {
    const bf16t* qp = Q + (size_t)(qs + ncol) * HD_;
    const int akb = half * 8;
    qa0.h[0] = *(const v8bf*)(qp + akb);
    qa0.h[1] = *(const v8bf*)(qp + 16 + akb);
    qa1.h[0] = *(const v8bf*)(qp + 32 + akb);
    qa1.h[1] = *(const v8bf*)(qp + 48 + akb);
  }

  const v8f vzero = {0.f,0.f,0.f,0.f,0.f,0.f,0.f,0.f};
  v8f o[4] = {vzero, vzero, vzero, vzero};
  float mr[8], lr[8];
#pragma unroll
  for (int r = 0; r < 8; ++r) { mr[r] = NEGINF; lr[r] = 0.f; }
  const int bkb = half * 8;

  for (int phase = 0; phase < 2; ++phase) {
    const bf16t* KB = phase ? Kw : Kc;
    const bf16t* VT = phase ? VwT : VcT;
    const int nk = phase ? L_ : LC_;
    const int LJ = phase ? L_ : LCP_;   // transposed-V row stride
    int kstart, kend;
    if (phase == 0) { kstart = 0; int kmax = (qs + 8) / STRIDE_; kend = kmax < (LC_ - 1) ? kmax : (LC_ - 1); }
    else            { kstart = (qs > (W_ - 1)) ? ((qs - (W_ - 1)) & ~31) : 0; kend = qs + 15; }

    for (int kb0 = kstart; kb0 <= kend; kb0 += 32) {
      // prefetch next tile's K rows toward the WGP (global_prefetch_b8)
      if (kb0 + 32 <= kend) {
        int pk = kb0 + 32 + lane; if (pk >= nk) pk = nk - 1;
        __builtin_prefetch(KB + (size_t)pk * HD_, 0, 3);
      }
      // ---- QK^T: preload 4 B-fragments, then 4 WMMAs ----
      ABFrag kb00, kb01, kb10, kb11;
      int key0 = kb0 + ncol;
      int key1 = kb0 + 16 + ncol;
      {
        int kc0 = key0 < nk ? key0 : nk - 1;
        int kc1 = key1 < nk ? key1 : nk - 1;
        const bf16t* kp0 = KB + (size_t)kc0 * HD_;
        const bf16t* kp1 = KB + (size_t)kc1 * HD_;
        kb00.h[0] = *(const v8bf*)(kp0 + bkb);
        kb00.h[1] = *(const v8bf*)(kp0 + bkb + 16);
        kb01.h[0] = *(const v8bf*)(kp0 + 32 + bkb);
        kb01.h[1] = *(const v8bf*)(kp0 + 32 + bkb + 16);
        kb10.h[0] = *(const v8bf*)(kp1 + bkb);
        kb10.h[1] = *(const v8bf*)(kp1 + bkb + 16);
        kb11.h[0] = *(const v8bf*)(kp1 + 32 + bkb);
        kb11.h[1] = *(const v8bf*)(kp1 + 32 + bkb + 16);
      }
      v8f s0 = vzero, s1 = vzero;
      s0 = wmma_bf16(qa0, kb00, s0);
      s0 = wmma_bf16(qa1, kb01, s0);
      s1 = wmma_bf16(qa0, kb10, s1);
      s1 = wmma_bf16(qa1, kb11, s1);

      float p0[8], p1[8];
#pragma unroll
      for (int r = 0; r < 8; ++r) {
        int q = qs + r + half * 8;
        bool v0, v1;
        if (phase == 0) {
          v0 = (key0 < LC_) && (q >= key0 * STRIDE_ + RATIO_ - 1);
          v1 = (key1 < LC_) && (q >= key1 * STRIDE_ + RATIO_ - 1);
        } else {
          int d0 = q - key0, d1 = q - key1;
          v0 = (d0 >= 0) && (d0 < W_);
          v1 = (d1 >= 0) && (d1 < W_);
        }
        p0[r] = v0 ? s0[r] * 0.125f : NEGINF;
        p1[r] = v1 ? s1[r] * 0.125f : NEGINF;
      }
      // online softmax update (half-wave shuffle reductions per row)
#pragma unroll
      for (int r = 0; r < 8; ++r) {
        float mv = fmaxf(p0[r], p1[r]);
        mv = fmaxf(mv, __shfl_xor(mv, 1, 32));
        mv = fmaxf(mv, __shfl_xor(mv, 2, 32));
        mv = fmaxf(mv, __shfl_xor(mv, 4, 32));
        mv = fmaxf(mv, __shfl_xor(mv, 8, 32));
        float mnew = fmaxf(mr[r], mv);
        float f, pe0, pe1;
        if (mnew < 0.5f * NEGINF) { f = 1.f; pe0 = 0.f; pe1 = 0.f; }
        else {
          f = __expf(mr[r] - mnew);
          pe0 = __expf(p0[r] - mnew);
          pe1 = __expf(p1[r] - mnew);
          mr[r] = mnew;
        }
        float rs = pe0 + pe1;
        rs += __shfl_xor(rs, 1, 32);
        rs += __shfl_xor(rs, 2, 32);
        rs += __shfl_xor(rs, 4, 32);
        rs += __shfl_xor(rs, 8, 32);
        lr[r] = lr[r] * f + rs;
#pragma unroll
        for (int nc = 0; nc < 4; ++nc) o[nc][r] *= f;
        pshare[wave][r + half * 8][ncol]      = f2b(pe0);
        pshare[wave][r + half * 8][16 + ncol] = f2b(pe1);
      }
      __builtin_amdgcn_wave_barrier();
      ABFrag pa;
      {
        const int akb = half * 8;
        const bf16t* pp = &pshare[wave][ncol][0];
        pa.h[0] = *(const v8bf*)(pp + akb);
        pa.h[1] = *(const v8bf*)(pp + 16 + akb);
      }
      // ---- PV: contiguous B-fragments from transposed V, then 4 WMMAs ----
      ABFrag vfr[4];
#pragma unroll
      for (int nc = 0; nc < 4; ++nc) {
        const bf16t* vp = VT + (size_t)(nc * 16 + ncol) * LJ + kb0 + bkb;
        vfr[nc].h[0] = *(const v8bf*)vp;
        vfr[nc].h[1] = *(const v8bf*)(vp + 16);
      }
#pragma unroll
      for (int nc = 0; nc < 4; ++nc) o[nc] = wmma_bf16(pa, vfr[nc], o[nc]);
      __builtin_amdgcn_wave_barrier();
    }
  }

  float snk = sinkl[h];
  const int b = bh >> 4;
#pragma unroll
  for (int r = 0; r < 8; ++r) {
    float m_ = mr[r], l_ = lr[r], corr;
    if (snk > m_) { corr = __expf(m_ - snk); l_ = l_ * corr + 1.f; }
    else          { corr = 1.f; l_ += __expf(snk - m_); }
    float inv = corr / l_;
    int q = qs + r + half * 8;
    size_t base = ((size_t)(b * L_ + q)) * D_ + (size_t)h * HD_;
#pragma unroll
    for (int nc = 0; nc < 4; ++nc)
      outb[base + nc * 16 + ncol] = f2b(o[nc][r] * inv);
  }
}

// ---------------- host launcher ----------------

extern "C" void kernel_launch(void* const* d_in, const int* in_sizes, int n_in,
                              void* d_out, int out_size, void* d_ws, size_t ws_size,
                              hipStream_t stream) {
  const float* x    = (const float*)d_in[0];
  const float* wq   = (const float*)d_in[1];
  const float* wk   = (const float*)d_in[2];
  const float* wv   = (const float*)d_in[3];
  const float* wo   = (const float*)d_in[4];
  const float* wkc  = (const float*)d_in[5];
  const float* wvc  = (const float*)d_in[6];
  const float* gate = (const float*)d_in[7];
  const float* sink = (const float*)d_in[8];
  float* out = (float*)d_out;
  (void)in_sizes; (void)n_in; (void)out_size; (void)ws_size;

  char* ws = (char*)d_ws;
  size_t off = 0;
  auto alloc = [&](size_t bytes) -> void* {
    void* p = ws + off;
    off = (off + bytes + 255) & ~(size_t)255;
    return p;
  };
  const size_t NX  = (size_t)B_ * L_ * D_;
  const size_t NW  = (size_t)D_ * D_;
  const size_t NXC = (size_t)B_ * LC_ * D_;
  const size_t NVC = (size_t)B_ * H_ * HD_ * LCP_;

  bf16t* x_bf   = (bf16t*)alloc(NX * 2);
  bf16t* wq_bf  = (bf16t*)alloc(NW * 2);
  bf16t* wk_bf  = (bf16t*)alloc(NW * 2);
  bf16t* wv_bf  = (bf16t*)alloc(NW * 2);
  bf16t* wo_bf  = (bf16t*)alloc(NW * 2);
  bf16t* wkc_bf = (bf16t*)alloc(NW * 2);
  bf16t* wvc_bf = (bf16t*)alloc(NW * 2);
  bf16t* xc_bf  = (bf16t*)alloc(NXC * 2);
  bf16t* q_bf   = (bf16t*)alloc(NX * 2);
  bf16t* k_bf   = (bf16t*)alloc(NX * 2);
  bf16t* v_bf   = (bf16t*)alloc(NX * 2);   // transposed [B,H,HD,L]
  bf16t* kc_bf  = (bf16t*)alloc(NXC * 2);
  bf16t* vc_bf  = (bf16t*)alloc(NVC * 2);  // transposed [B,H,HD,LCP]
  bf16t* at_bf  = (bf16t*)alloc(NX * 2);
  float* scratch = (float*)alloc(NX * 4);

  const int T = 256;
  cvt_bf16_kernel<<<(int)((NX + T - 1) / T), T, 0, stream>>>(x,   x_bf,   (int)NX);
  cvt_bf16_kernel<<<(int)((NW + T - 1) / T), T, 0, stream>>>(wq,  wq_bf,  (int)NW);
  cvt_bf16_kernel<<<(int)((NW + T - 1) / T), T, 0, stream>>>(wk,  wk_bf,  (int)NW);
  cvt_bf16_kernel<<<(int)((NW + T - 1) / T), T, 0, stream>>>(wv,  wv_bf,  (int)NW);
  cvt_bf16_kernel<<<(int)((NW + T - 1) / T), T, 0, stream>>>(wo,  wo_bf,  (int)NW);
  cvt_bf16_kernel<<<(int)((NW + T - 1) / T), T, 0, stream>>>(wkc, wkc_bf, (int)NW);
  cvt_bf16_kernel<<<(int)((NW + T - 1) / T), T, 0, stream>>>(wvc, wvc_bf, (int)NW);

  const dim3 gdimFull(D_ / 256, (B_ * L_) / 32);
  const dim3 gdimComp(D_ / 256, (B_ * LC_ + 31) / 32);
  const int ropeN = B_ * L_ * H_ * (HD_ / 2);

  gemm_nt_bf16<<<gdimFull, 128, 0, stream>>>(x_bf, wq_bf, scratch, B_ * L_, D_, D_);
  rope_transpose_kernel<<<(ropeN + T - 1) / T, T, 0, stream>>>(scratch, q_bf);
  gemm_nt_bf16<<<gdimFull, 128, 0, stream>>>(x_bf, wk_bf, scratch, B_ * L_, D_, D_);
  rope_transpose_kernel<<<(ropeN + T - 1) / T, T, 0, stream>>>(scratch, k_bf);
  gemm_nt_bf16<<<gdimFull, 128, 0, stream>>>(x_bf, wv_bf, scratch, B_ * L_, D_, D_);
  cvt_transpose_vT_kernel<<<(int)((NX + T - 1) / T), T, 0, stream>>>(scratch, v_bf, L_, L_);
  compress_kernel<<<(int)((NXC + T - 1) / T), T, 0, stream>>>(x, gate, xc_bf);
  gemm_nt_bf16<<<gdimComp, 128, 0, stream>>>(xc_bf, wkc_bf, scratch, B_ * LC_, D_, D_);
  cvt_transpose_kernel<<<(int)((NXC + T - 1) / T), T, 0, stream>>>(scratch, kc_bf, LC_);
  gemm_nt_bf16<<<gdimComp, 128, 0, stream>>>(xc_bf, wvc_bf, scratch, B_ * LC_, D_, D_);
  cvt_transpose_vT_kernel<<<(int)((NVC + T - 1) / T), T, 0, stream>>>(scratch, vc_bf, LC_, LCP_);
  attn_kernel<<<B_ * H_ * (L_ / 64), 128, 0, stream>>>(q_bf, k_bf, v_bf, kc_bf, vc_bf, sink, at_bf);
  gemm_nt_bf16<<<gdimFull, 128, 0, stream>>>(at_bf, wo_bf, out, B_ * L_, D_, D_);
}
